// ASTCRFCell_197568495725
// MI455X (gfx1250) — matrix operand order repaired and checked by reading, not compiled, not run on hardware
//
#include <hip/hip_runtime.h>

// ---------------------------------------------------------------------------
// ASTCRF cell for MI455X (gfx1250, wave32, WMMA bf16 16x16x32).
// Dominant cost: masked-HHT @ weight GEMMs (~550 GFLOP) -> bf16 WMMA.
// This revision: fragment-friendly LDS layouts (transposed B tiles) so every
// WMMA operand is fed by ds_load_b128 instead of ds_load_u16 gathers.
// ---------------------------------------------------------------------------

#define NODE  2048
#define NBATCH 16
#define EMB   16
#define DIN   66     // 2 + 64
#define DOUT  64
#define GOUT  128    // 2*DOUT
#define KCAT  132    // CHEB_K * DIN
#define KPAD  160    // KCAT padded to multiple of 32
#define FPAD  128    // feature dim padded for GEMM

typedef __bf16 bf16_t;
typedef bf16_t v16bf __attribute__((ext_vector_type(16)));
typedef bf16_t v8bf  __attribute__((ext_vector_type(8)));
typedef float  v8f   __attribute__((ext_vector_type(8)));

// ---------------- WMMA helpers (layouts per CDNA5 ISA 7.12.2) ----------------

__device__ __forceinline__ v8f wmma_bf16(v16bf a, v16bf b, v8f c) {
  return __builtin_amdgcn_wmma_f32_16x16x32_bf16(false, a, false, b,
                                                 (short)0, c, false, false);
}

// A fragment from row-major [16][ld] bf16 tile (ld multiple of 8, 16B aligned).
// lane L(0-15): M=L, K=0..7,16..23 ; lane L(16-31): M=L-16, K=8..15,24..31.
__device__ __forceinline__ v16bf load_a_frag(const bf16_t* p, int ld, int lane) {
  const bf16_t* row = p + (lane & 15) * ld + ((lane >> 4) << 3);
  v8bf lo = *(const v8bf*)(row);
  v8bf hi = *(const v8bf*)(row + 16);
  return __builtin_shufflevector(lo, hi, 0, 1, 2, 3, 4, 5, 6, 7,
                                         8, 9, 10, 11, 12, 13, 14, 15);
}

// B fragment from TRANSPOSED [16 cols][ldk] bf16 tile.
// lane L(0-15): N=L, K=0..7,16..23 ; lane L(16-31): N=L-16, K=8..15,24..31.
__device__ __forceinline__ v16bf load_bT_frag(const bf16_t* p, int ldk, int lane) {
  const bf16_t* col = p + (lane & 15) * ldk + ((lane >> 4) << 3);
  v8bf lo = *(const v8bf*)(col);
  v8bf hi = *(const v8bf*)(col + 16);
  return __builtin_shufflevector(lo, hi, 0, 1, 2, 3, 4, 5, 6, 7,
                                         8, 9, 10, 11, 12, 13, 14, 15);
}

// ---------------- small prep kernels ----------------

// A[n,m] = dot(E[n,:], E[m,:])  (raw signed similarity, reused for masks)
__global__ void __launch_bounds__(256)
astcrf_gram(const float* __restrict__ E, float* __restrict__ Amat) {
  __shared__ float er[EMB];
  const int n = blockIdx.y;
  const int m = blockIdx.x * 256 + threadIdx.x;
  if (threadIdx.x < EMB) er[threadIdx.x] = E[n * EMB + threadIdx.x];
  __syncthreads();
  if (m < NODE) {
    float s = 0.f;
#pragma unroll
    for (int e = 0; e < EMB; ++e) s += er[e] * E[m * EMB + e];
    Amat[(long long)n * NODE + m] = s;
  }
}

// row softmax (optionally after relu), f32 in -> bf16 out
__global__ void __launch_bounds__(256)
astcrf_row_softmax(const float* __restrict__ in, bf16_t* __restrict__ out,
                   int ncols, int doRelu) {
  const int tid = threadIdx.x;
  const long long row = blockIdx.x;
  in  += row * ncols;
  out += row * ncols;
  __shared__ float red[256];
  float m = -3.0e38f;
  for (int c = tid; c < ncols; c += 256) {
    float v = in[c];
    if (doRelu) v = fmaxf(v, 0.f);
    m = fmaxf(m, v);
  }
  red[tid] = m; __syncthreads();
  for (int s2 = 128; s2 > 0; s2 >>= 1) {
    if (tid < s2) red[tid] = fmaxf(red[tid], red[tid + s2]);
    __syncthreads();
  }
  m = red[0]; __syncthreads();
  float s = 0.f;
  for (int c = tid; c < ncols; c += 256) {
    float v = in[c];
    if (doRelu) v = fmaxf(v, 0.f);
    s += __expf(v - m);
  }
  red[tid] = s; __syncthreads();
  for (int s2 = 128; s2 > 0; s2 >>= 1) {
    if (tid < s2) red[tid] += red[tid + s2];
    __syncthreads();
  }
  const float inv = 1.f / red[0];
  for (int c = tid; c < ncols; c += 256) {
    float v = in[c];
    if (doRelu) v = fmaxf(v, 0.f);
    out[c] = (bf16_t)(__expf(v - m) * inv);
  }
}

__global__ void __launch_bounds__(256)
astcrf_cvt_bf16(const float* __restrict__ in, bf16_t* __restrict__ out,
                long long count) {
  long long i = (long long)blockIdx.x * 256 + threadIdx.x;
  if (i < count) out[i] = (bf16_t)in[i];
}

// inp_bf[b,n,f] = f<2 ? x : f<66 ? state : 0   ([B,N,FPAD] bf16)
__global__ void __launch_bounds__(256)
astcrf_pack_inp(const float* __restrict__ x, const float* __restrict__ st,
                bf16_t* __restrict__ out) {
  long long i = (long long)blockIdx.x * 256 + threadIdx.x;
  if (i >= (long long)NBATCH * NODE * FPAD) return;
  int f = (int)(i & (FPAD - 1));
  long long bn = i >> 7;
  float v = (f < 2) ? x[bn * 2 + f] : (f < DIN ? st[bn * DOUT + (f - 2)] : 0.f);
  out[i] = (bf16_t)v;
}

// ---------------- generic bf16 WMMA GEMM: C[MxN] = A[MxK] * B[KxN] ----------
// mode 0: C = acc (f32)
// mode 1: Cb = bf16( acc * signmask(maskSrc) )   (fused adj_mask * HHT)
// mode 2: C += alpha * acc                        (fused deg @ h accumulation)
__global__ void __launch_bounds__(256)
astcrf_gemm(const bf16_t* __restrict__ A, long long sA, int lda,
            const bf16_t* __restrict__ B, long long sB, int ldb,
            float* __restrict__ C, long long sC, int ldc,
            int M, int N, int K, int mode, float alpha,
            const float* __restrict__ maskSrc, int ldm, int maskSel,
            bf16_t* __restrict__ Cb, int ldcb) {
  constexpr int BM = 128, BN = 128, BK = 32;
  __shared__ alignas(16) bf16_t As[BM * BK];   // row-major [BM][BK]
  __shared__ alignas(16) bf16_t Bst[BN * BK];  // TRANSPOSED [BN][BK]
  const int tid  = threadIdx.x;
  const int lane = tid & 31;
  const int wave = tid >> 5;
  const long long bz = blockIdx.z;
  A += bz * sA; B += bz * sB; C += bz * sC;
  const int row0 = blockIdx.y * BM;
  const int col0 = blockIdx.x * BN;
  const int wm = (wave & 3) * 32;   // 2 M-tiles per wave
  const int wn = (wave >> 2) * 64;  // 4 N-tiles per wave
  const bool full = (row0 + BM <= M) && (col0 + BN <= N) && ((K & (BK - 1)) == 0);

  v8f acc[2][4] = {};

  for (int k0 = 0; k0 < K; k0 += BK) {
    if (full) {
      // A: 8 contiguous bf16 per thread x2 (b128 global loads, b128 LDS stores)
#pragma unroll
      for (int t = 0; t < 2; ++t) {
        int i = (tid + t * 256) << 3;          // [0, 4096)
        int r = i >> 5, c = i & 31;
        *(v8bf*)(As + i) =
            *(const v8bf*)(A + (long long)(row0 + r) * lda + k0 + c);
      }
      // B: coalesced 8-wide global read, transposed scatter into LDS
#pragma unroll
      for (int t = 0; t < 2; ++t) {
        int i = (tid + t * 256) << 3;          // [0, 4096)
        int r = i >> 7, c = i & 127;           // r = k row, c = column
        v8bf v = *(const v8bf*)(B + (long long)(k0 + r) * ldb + col0 + c);
#pragma unroll
        for (int e = 0; e < 8; ++e) Bst[(c + e) * BK + r] = v[e];
      }
    } else {
      for (int i = tid; i < BM * BK; i += 256) {
        int r = i >> 5, c = i & 31;
        int gm = row0 + r, gk = k0 + c;
        As[i] = (gm < M && gk < K) ? A[(long long)gm * lda + gk] : (bf16_t)0.0f;
      }
      for (int i = tid; i < BK * BN; i += 256) {
        int r = i >> 7, c = i & 127;
        int gk = k0 + r, gn = col0 + c;
        Bst[c * BK + r] =
            (gk < K && gn < N) ? B[(long long)gk * ldb + gn] : (bf16_t)0.0f;
      }
    }
    if (k0 + BK < K) {  // pull next K-panel toward L2 (global_prefetch_b8)
      __builtin_prefetch(&A[(long long)(row0 + (tid & 127)) * lda + k0 + BK], 0, 1);
      __builtin_prefetch(&B[(long long)(k0 + BK + (tid & 31)) * ldb + col0], 0, 1);
    }
    __syncthreads();

    v16bf af[2], bfr[4];
#pragma unroll
    for (int mi = 0; mi < 2; ++mi)
      af[mi] = load_a_frag(As + (wm + mi * 16) * BK, BK, lane);
#pragma unroll
    for (int ni = 0; ni < 4; ++ni)
      bfr[ni] = load_bT_frag(Bst + (wn + ni * 16) * BK, BK, lane);
#pragma unroll
    for (int mi = 0; mi < 2; ++mi)
#pragma unroll
      for (int ni = 0; ni < 4; ++ni)
        acc[mi][ni] = wmma_bf16(af[mi], bfr[ni], acc[mi][ni]);
    __syncthreads();
  }

#pragma unroll
  for (int mi = 0; mi < 2; ++mi) {
#pragma unroll
    for (int ni = 0; ni < 4; ++ni) {
      int gn = col0 + wn + ni * 16 + (lane & 15);
      int mbase = row0 + wm + mi * 16 + ((lane >> 4) << 3);
#pragma unroll
      for (int rr = 0; rr < 8; ++rr) {
        int gm = mbase + rr;
        if (gm < M && gn < N) {
          float v = acc[mi][ni][rr];
          if (mode == 0) {
            C[(long long)gm * ldc + gn] = v;
          } else if (mode == 1) {
            float s = maskSrc[(long long)gm * ldm + gn];
            bool keep = (maskSel > 0) ? (s > 0.f) : (s < 0.f);
            Cb[(long long)gm * ldcb + gn] = (bf16_t)(keep ? v : 0.f);
          } else {
            C[(long long)gm * ldc + gn] += alpha * v;
          }
        }
      }
    }
  }
}

// ---------------- per-node adaptive-weight GEMMs (M = 16 batches) ----------

// gate: z_r = sigmoid([inp, sup@inp] @ Wn + bn); write r, cand=[x, z*state]
__global__ void __launch_bounds__(128)
astcrf_gate_node(const float* __restrict__ E,  const float* __restrict__ gw,
                 const float* __restrict__ gb, const bf16_t* __restrict__ inpb,
                 const float* __restrict__ xg, const float* __restrict__ x,
                 const float* __restrict__ st, float* __restrict__ r_out,
                 bf16_t* __restrict__ candb) {
  const int n = blockIdx.x;
  const int tid = threadIdx.x, lane = tid & 31, wave = tid >> 5;
  __shared__ float er[EMB];
  __shared__ float bn[GOUT];
  __shared__ alignas(16) bf16_t WnT[GOUT * KPAD];  // TRANSPOSED [o][ki], 40 KB
  __shared__ alignas(16) bf16_t Xc[16 * KPAD];     // [batch][ki], 5 KB
  if (tid < EMB) er[tid] = E[n * EMB + tid];
  __syncthreads();
  for (int idx = tid; idx < KCAT * GOUT; idx += 128) {  // Wn = sum_e E[n,e]*gw[e]
    int ki = idx >> 7, o = idx & 127;                   // coalesced gw reads
    float s = 0.f;
#pragma unroll
    for (int e = 0; e < EMB; ++e) s += er[e] * gw[((e * KCAT) + ki) * GOUT + o];
    WnT[o * KPAD + ki] = (bf16_t)s;
  }
  for (int idx = tid; idx < (KPAD - KCAT) * GOUT; idx += 128) {
    int ki = KCAT + (idx >> 7), o = idx & 127;
    WnT[o * KPAD + ki] = (bf16_t)0.f;
  }
  {
    float s = 0.f;
#pragma unroll
    for (int e = 0; e < EMB; ++e) s += er[e] * gb[e * GOUT + tid];
    bn[tid] = s;  // 128 threads == GOUT
  }
  for (int idx = tid; idx < 16 * KPAD; idx += 128) {     // Xc = [inp, sup@inp]
    int b = idx / KPAD, ki = idx % KPAD;
    long long base = (long long)b * NODE + n;
    bf16_t v;
    if (ki < DIN)       v = inpb[base * FPAD + ki];
    else if (ki < KCAT) v = (bf16_t)xg[base * FPAD + (ki - DIN)];
    else                v = (bf16_t)0.f;
    Xc[b * KPAD + ki] = v;
  }
  __syncthreads();

  v8f acc[2] = {};
  for (int kk = 0; kk < KPAD; kk += 32) {
    v16bf af = load_a_frag(Xc + kk, KPAD, lane);
#pragma unroll
    for (int t = 0; t < 2; ++t) {
      int col = (wave * 2 + t) * 16;
      v16bf bfrag = load_bT_frag(WnT + col * KPAD + kk, KPAD, lane);
      acc[t] = wmma_bf16(af, bfrag, acc[t]);
    }
  }
#pragma unroll
  for (int t = 0; t < 2; ++t) {
    int o = (wave * 2 + t) * 16 + (lane & 15);
#pragma unroll
    for (int rr = 0; rr < 8; ++rr) {
      int b = ((lane >> 4) << 3) + rr;
      long long base = (long long)b * NODE + n;
      float v = 1.f / (1.f + __expf(-(acc[t][rr] + bn[o])));
      if (o < DOUT)  // z
        candb[base * FPAD + 2 + o] = (bf16_t)(v * st[base * DOUT + o]);
      else           // r
        r_out[base * DOUT + (o - DOUT)] = v;
    }
  }
  for (int idx = tid; idx < 16 * 2; idx += 128) {          // cand[:,:,0:2] = x
    int b = idx >> 1, f = idx & 1;
    candb[((long long)b * NODE + n) * FPAD + f] =
        (bf16_t)x[((long long)b * NODE + n) * 2 + f];
  }
  for (int idx = tid; idx < 16 * (FPAD - DIN); idx += 128) {  // pad = 0
    int b = idx / (FPAD - DIN), f = DIN + idx % (FPAD - DIN);
    candb[((long long)b * NODE + n) * FPAD + f] = (bf16_t)0.f;
  }
}

// update: hc = tanh([cand, sup@cand] @ Wn + bn); h = r*state + (1-r)*hc
__global__ void __launch_bounds__(128)
astcrf_upd_node(const float* __restrict__ E,  const float* __restrict__ uw,
                const float* __restrict__ ub, const bf16_t* __restrict__ candb,
                const float* __restrict__ xg, const float* __restrict__ st,
                const float* __restrict__ r_in, float* __restrict__ h,
                bf16_t* __restrict__ hb, bf16_t* __restrict__ hTb,
                float* __restrict__ out) {
  const int n = blockIdx.x;
  const int tid = threadIdx.x, lane = tid & 31, wave = tid >> 5;
  __shared__ float er[EMB];
  __shared__ float bn[DOUT];
  __shared__ alignas(16) bf16_t WnT[DOUT * KPAD];  // TRANSPOSED [o][ki], 20 KB
  __shared__ alignas(16) bf16_t Xc[16 * KPAD];
  if (tid < EMB) er[tid] = E[n * EMB + tid];
  __syncthreads();
  for (int idx = tid; idx < KCAT * DOUT; idx += 128) {
    int ki = idx >> 6, o = idx & 63;
    float s = 0.f;
#pragma unroll
    for (int e = 0; e < EMB; ++e) s += er[e] * uw[((e * KCAT) + ki) * DOUT + o];
    WnT[o * KPAD + ki] = (bf16_t)s;
  }
  for (int idx = tid; idx < (KPAD - KCAT) * DOUT; idx += 128) {
    int ki = KCAT + (idx >> 6), o = idx & 63;
    WnT[o * KPAD + ki] = (bf16_t)0.f;
  }
  if (tid < DOUT) {
    float s = 0.f;
#pragma unroll
    for (int e = 0; e < EMB; ++e) s += er[e] * ub[e * DOUT + tid];
    bn[tid] = s;
  }
  for (int idx = tid; idx < 16 * KPAD; idx += 128) {
    int b = idx / KPAD, ki = idx % KPAD;
    long long base = (long long)b * NODE + n;
    bf16_t v;
    if (ki < DIN)       v = candb[base * FPAD + ki];
    else if (ki < KCAT) v = (bf16_t)xg[base * FPAD + (ki - DIN)];
    else                v = (bf16_t)0.f;
    Xc[b * KPAD + ki] = v;
  }
  __syncthreads();

  v8f acc = {};
  for (int kk = 0; kk < KPAD; kk += 32) {
    v16bf af = load_a_frag(Xc + kk, KPAD, lane);
    v16bf bfrag = load_bT_frag(WnT + (wave * 16) * KPAD + kk, KPAD, lane);
    acc = wmma_bf16(af, bfrag, acc);
  }
  int o = wave * 16 + (lane & 15);
#pragma unroll
  for (int rr = 0; rr < 8; ++rr) {
    int b = ((lane >> 4) << 3) + rr;
    long long base = (long long)b * NODE + n;
    float hc = tanhf(acc[rr] + bn[o]);
    float rv = r_in[base * DOUT + o];
    float hv = rv * st[base * DOUT + o] + (1.f - rv) * hc;
    h[base * DOUT + o]  = hv;
    hb[base * DOUT + o] = (bf16_t)hv;
    hTb[((long long)b * DOUT + o) * NODE + n] = (bf16_t)hv;   // transposed copy
    out[base * DOUT + o] = 0.8f * hv;                          // init numerator
  }
}

// ---------------------------------------------------------------------------

extern "C" void kernel_launch(void* const* d_in, const int* in_sizes, int n_in,
                              void* d_out, int out_size, void* d_ws, size_t ws_size,
                              hipStream_t stream) {
  (void)in_sizes; (void)n_in; (void)out_size; (void)ws_size;
  const float* x  = (const float*)d_in[0];
  const float* st = (const float*)d_in[1];
  const float* E  = (const float*)d_in[2];
  const float* wc = (const float*)d_in[3];
  const float* wd = (const float*)d_in[4];
  const float* gw = (const float*)d_in[5];
  const float* gb = (const float*)d_in[6];
  const float* uw = (const float*)d_in[7];
  const float* ub = (const float*)d_in[8];
  float* out = (float*)d_out;

  char* ws = (char*)d_ws;
  size_t off = 0;
  auto alloc = [&](size_t bytes) -> void* {
    void* p = ws + off;
    off += (bytes + 255) & ~(size_t)255;
    return p;
  };
  float*  Amat  = (float*) alloc((size_t)NODE * NODE * 4);
  bf16_t* supb  = (bf16_t*)alloc((size_t)NODE * NODE * 2);
  bf16_t* inpb  = (bf16_t*)alloc((size_t)NBATCH * NODE * FPAD * 2);
  bf16_t* candb = (bf16_t*)alloc((size_t)NBATCH * NODE * FPAD * 2);
  float*  xg    = (float*) alloc((size_t)NBATCH * NODE * FPAD * 4);
  float*  rbuf  = (float*) alloc((size_t)NBATCH * NODE * DOUT * 4);
  float*  h     = (float*) alloc((size_t)NBATCH * NODE * DOUT * 4);
  bf16_t* hb    = (bf16_t*)alloc((size_t)NBATCH * NODE * DOUT * 2);
  bf16_t* hTb   = (bf16_t*)alloc((size_t)NBATCH * DOUT * NODE * 2);
  bf16_t* wcb   = (bf16_t*)alloc((size_t)NODE * NODE * 2);
  bf16_t* wdb   = (bf16_t*)alloc((size_t)NODE * NODE * 2);
  bf16_t* Pb    = (bf16_t*)alloc((size_t)NODE * NODE * 2);
  float*  T     = (float*) alloc((size_t)NODE * NODE * 4);

  auto gemm = [&](const bf16_t* Ap, long long sAp, int lda,
                  const bf16_t* Bp, long long sBp, int ldb,
                  float* Cp, long long sCp, int ldc,
                  int M, int Nn, int K, int batches, int mode, float alpha,
                  const float* maskSrc, int ldm, int maskSel,
                  bf16_t* Cbp, int ldcb) {
    dim3 grid((Nn + 127) / 128, (M + 127) / 128, batches);
    astcrf_gemm<<<grid, dim3(256), 0, stream>>>(
        Ap, sAp, lda, Bp, sBp, ldb, Cp, sCp, ldc, M, Nn, K,
        mode, alpha, maskSrc, ldm, maskSel, Cbp, ldcb);
  };

  // 1) A = E E^T ; supports = softmax(relu(A)) (bf16)
  astcrf_gram<<<dim3(NODE / 256, NODE), 256, 0, stream>>>(E, Amat);
  astcrf_row_softmax<<<NODE, 256, 0, stream>>>(Amat, supb, NODE, 1);

  // 2) weights -> bf16
  const long long wn2 = (long long)NODE * NODE;
  astcrf_cvt_bf16<<<(int)((wn2 + 255) / 256), 256, 0, stream>>>(wc, wcb, wn2);
  astcrf_cvt_bf16<<<(int)((wn2 + 255) / 256), 256, 0, stream>>>(wd, wdb, wn2);

  // 3) inp = [x, state] (bf16, padded)
  const long long icnt = (long long)NBATCH * NODE * FPAD;
  astcrf_pack_inp<<<(int)((icnt + 255) / 256), 256, 0, stream>>>(x, st, inpb);

  // 4) xg = supports @ inp  (batched over z)
  gemm(supb, 0, NODE, inpb, (long long)NODE * FPAD, FPAD,
       xg, (long long)NODE * FPAD, FPAD, NODE, FPAD, NODE, NBATCH,
       0, 0.f, Amat, NODE, 0, Pb, NODE);

  // 5) gate (per-node adaptive weights, fused sigmoid / z*state / r)
  astcrf_gate_node<<<NODE, 128, 0, stream>>>(E, gw, gb, inpb, xg, x, st,
                                             rbuf, candb);

  // 6) xg = supports @ cand
  gemm(supb, 0, NODE, candb, (long long)NODE * FPAD, FPAD,
       xg, (long long)NODE * FPAD, FPAD, NODE, FPAD, NODE, NBATCH,
       0, 0.f, Amat, NODE, 0, Pb, NODE);

  // 7) update (fused tanh + GRU mix; writes h, h_bf, h^T_bf, out = 0.8*h)
  astcrf_upd_node<<<NODE, 128, 0, stream>>>(E, uw, ub, candb, xg, st, rbuf,
                                            h, hb, hTb, out);

  // 8) degrees: per batch, per sign-mask — the 550 GF bulk of the workload
  for (int b = 0; b < NBATCH; ++b) {
    const bf16_t* hb_b  = hb  + (long long)b * NODE * DOUT;
    const bf16_t* hTb_b = hTb + (long long)b * DOUT * NODE;
    float* out_b = out + (long long)b * NODE * DOUT;
    for (int pass = 0; pass < 2; ++pass) {
      // P = mask(A) * (h h^T)  -> bf16
      gemm(hb_b, 0, DOUT, hTb_b, 0, NODE, T, 0, NODE,
           NODE, NODE, DOUT, 1, 1, 0.f, Amat, NODE, pass == 0 ? 1 : -1,
           Pb, NODE);
      // T = P @ W   (dominant GEMM: 2048^3, bf16 WMMA, L2-resident operands)
      gemm(Pb, 0, NODE, pass == 0 ? wcb : wdb, 0, NODE, T, 0, NODE,
           NODE, NODE, NODE, 1, 0, 0.f, Amat, NODE, 0, Pb, NODE);
      // deg = softmax rows of T (reuse Pb)
      astcrf_row_softmax<<<NODE, 256, 0, stream>>>(T, Pb, NODE, 0);
      // out += (+/-0.1) * deg @ h
      gemm(Pb, 0, NODE, hb_b, 0, DOUT, out_b, 0, DOUT,
           NODE, DOUT, NODE, 1, 2, pass == 0 ? -0.1f : 0.1f,
           Amat, NODE, 0, Pb, NODE);
    }
  }
}
